// OptimizedALiBiMultiHeadAttention_84731114816122
// MI455X (gfx1250) — matrix operand order repaired and checked by reading, not compiled, not run on hardware
//
#include <hip/hip_runtime.h>

typedef __attribute__((ext_vector_type(16))) __bf16 v16bf;
typedef __attribute__((ext_vector_type(8)))  __bf16 v8bf;
typedef __attribute__((ext_vector_type(8)))  float  v8f;

union BF16Frag {
  v16bf v;
  v8bf  h[2];
  unsigned short u[16];
};

__device__ __forceinline__ unsigned short f2bf(float f) {
  union { float f; unsigned int u; } c; c.f = f;
  unsigned int u = c.u;
  unsigned int r = (u + 0x7FFFu + ((u >> 16) & 1u)) >> 16;  // RNE
  return (unsigned short)r;
}

// ---------------------------------------------------------------------------
// Kernel 1: x [8192 x 1024] fp32 -> bf16 (row-major, A-fragment friendly)
// ---------------------------------------------------------------------------
__global__ void cvt_x_kernel(const float* __restrict__ x,
                             unsigned short* __restrict__ xb, int n) {
  int i = blockIdx.x * blockDim.x + threadIdx.x;
  if (i < n) xb[i] = f2bf(x[i]);
}

// ---------------------------------------------------------------------------
// Kernel 2: W_kqv [1024 x 3072] fp32 -> bf16 transposed [3072 x 1024]
// so a WMMA B-fragment (16 contiguous K values per lane) is one 32B load.
// ---------------------------------------------------------------------------
__global__ void cvt_wt_kernel(const float* __restrict__ wsrc,
                              unsigned short* __restrict__ wt) {
  int i = blockIdx.x * blockDim.x + threadIdx.x;   // i over K*N, coalesced read
  int k = i / 3072;
  int n = i % 3072;
  wt[(size_t)n * 1024 + k] = f2bf(wsrc[i]);
}

// ---------------------------------------------------------------------------
// Kernel 3: KQV GEMM. M=8192, N=3072, K=1024 in bf16, fp32 accum via WMMA.
// 8 waves (2x4), block tile 128x256, wave tile 64x64 (4x4 accs of v8f):
// 0.5 KB of fragments per WMMA -> minimizes L2 fragment traffic.
// Epilogue: cols [0,1024)=K -> kq, [1024,2048)=Q*(1/32) -> kq,
//           [2048,3072)=V -> transposed Vt[bh][d][s].
// ---------------------------------------------------------------------------
__global__ __launch_bounds__(256) void kqv_gemm_kernel(
    const unsigned short* __restrict__ xb,   // [8192 x 1024]
    const unsigned short* __restrict__ wt,   // [3072 x 1024] N-major
    unsigned short* __restrict__ kq,         // [8192 x 2048]
    unsigned short* __restrict__ vt)         // [64 x 64 x 2048]
{
  const int lane = threadIdx.x & 31;
  const int w    = threadIdx.x >> 5;
  const int wm = w & 1, wn = w >> 1;
  const int m0 = blockIdx.y * 128 + wm * 64;
  const int n0 = blockIdx.x * 256 + wn * 64;
  const int lr = lane & 15;
  const int hi = (lane < 16) ? 0 : 1;

  v8f acc[4][4];
  #pragma unroll
  for (int i = 0; i < 4; i++)
    #pragma unroll
    for (int j = 0; j < 4; j++) acc[i][j] = {};

  for (int kc = 0; kc < 1024; kc += 32) {
    BF16Frag a[4];
    #pragma unroll
    for (int i = 0; i < 4; i++) {
      // A fragment, contiguous k order (same as B): lanes<16 K=0..15, lanes>=16 K=16..31
      const unsigned short* ap = xb + (size_t)(m0 + i * 16 + lr) * 1024 + kc + hi * 16;
      a[i].h[0] = *(const v8bf*)(ap);
      a[i].h[1] = *(const v8bf*)(ap + 8);
    }
    BF16Frag bb[4];
    #pragma unroll
    for (int j = 0; j < 4; j++) {
      // B fragment, contiguous k order (same as A): col = lane%16; lanes<16 K=0..15, lanes>=16 K=16..31
      const unsigned short* bp = wt + (size_t)(n0 + j * 16 + lr) * 1024 + kc + hi * 16;
      bb[j].v = *(const v16bf*)bp;
    }
    #pragma unroll
    for (int i = 0; i < 4; i++)
      #pragma unroll
      for (int j = 0; j < 4; j++)
        acc[i][j] = __builtin_amdgcn_wmma_f32_16x16x32_bf16(
            false, a[i].v, false, bb[j].v, (short)0, acc[i][j], false, false);
  }

  // Epilogue. C layout: VGPR r -> rows r (lanes 0-15) / r+8 (lanes 16-31), N = lane%16
  #pragma unroll
  for (int i = 0; i < 4; i++)
    #pragma unroll
    for (int j = 0; j < 4; j++)
      #pragma unroll
      for (int r = 0; r < 8; r++) {
        int row = m0 + i * 16 + r + hi * 8;
        int n   = n0 + j * 16 + lr;
        float val = acc[i][j][r];
        if (n < 1024) {                       // K chunk (source order: k,q,v)
          kq[(size_t)row * 2048 + n] = f2bf(val);
        } else if (n < 2048) {                // Q chunk, pre-scaled by 1/sqrt(D)=1/32
          kq[(size_t)row * 2048 + n] = f2bf(val * 0.03125f);
        } else {                              // V chunk -> transposed Vt[bh][d][s]
          int hh = (n - 2048) >> 6, d = (n - 2048) & 63;
          int b = row >> 11, s = row & 2047;
          vt[((size_t)(b * 16 + hh) * 64 + d) * 2048 + s] = f2bf(val);
        }
      }
}

// ---------------------------------------------------------------------------
// Kernel 4: fused ALiBi attention (no causal mask; full-row online softmax).
// One wave per 32-row Q tile (two 16-row WMMA tiles sharing each K/V fragment);
// 8 waves/block; key blocks of 32.
// ---------------------------------------------------------------------------
__global__ __launch_bounds__(256) void alibi_attn_kernel(
    const unsigned short* __restrict__ kq,   // [8192 x 2048] (K | Q*1/32)
    const unsigned short* __restrict__ vt,   // [64 x 64 x 2048]
    float* __restrict__ out)                 // [4 x 2048 x 1024] fp32
{
  __shared__ unsigned short ptile[8][32][32];  // per-wave P transpose buffer (16KB)

  const int lane   = threadIdx.x & 31;
  const int w      = threadIdx.x >> 5;
  const int lr     = lane & 15;
  const int hiHalf = (lane < 16) ? 0 : 1;
  const int bh = blockIdx.y;
  const int b  = bh >> 4, h = bh & 15;
  const int q0 = blockIdx.x * 256 + w * 32;

  // ALiBi slope: 1/((2^8)^((h+1)/16)) = 2^(-(h+1)/2)
  const float slope = exp2f(-0.5f * (float)(h + 1));

  // Q A-fragments: row tiles i in {0,1}, d-chunks c in {[0,32),[32,64)}
  BF16Frag qa[2][2];
  #pragma unroll
  for (int i = 0; i < 2; i++) {
    const unsigned short* base =
        kq + ((size_t)(b * 2048 + q0 + i * 16 + lr)) * 2048 + 1024 + h * 64;
    #pragma unroll
    for (int c = 0; c < 2; c++) {
      const unsigned short* p = base + c * 32 + hiHalf * 16;
      qa[i][c].h[0] = *(const v8bf*)(p);
      qa[i][c].h[1] = *(const v8bf*)(p + 8);
    }
  }

  float mrow[2][8], lrow[2][8];
  #pragma unroll
  for (int i = 0; i < 2; i++)
    #pragma unroll
    for (int r = 0; r < 8; r++) { mrow[i][r] = -1e30f; lrow[i][r] = 0.0f; }
  v8f o[2][4];
  #pragma unroll
  for (int i = 0; i < 2; i++)
    #pragma unroll
    for (int c = 0; c < 4; c++) o[i][c] = {};

  const unsigned short* kbase = kq + (size_t)b * 2048 * 2048 + h * 64;
  const unsigned short* vbase = vt + (size_t)bh * 64 * 2048;

  for (int j0 = 0; j0 < 2048; j0 += 32) {
    // ---- K B-fragments, shared by both Q row tiles ----
    BF16Frag kb[2][2];   // [key half t][d chunk]
    #pragma unroll
    for (int t = 0; t < 2; t++) {
      const unsigned short* kp = kbase + (size_t)(j0 + t * 16 + lr) * 2048;
      kb[t][0].v = *(const v16bf*)(kp + hiHalf * 16);        // d 0..31
      kb[t][1].v = *(const v16bf*)(kp + 32 + hiHalf * 16);   // d 32..63
    }
    // ---- scores: S[i][16x32] = Q_i[16x64] . K^T ----
    v8f s[2][2];
    #pragma unroll
    for (int i = 0; i < 2; i++)
      #pragma unroll
      for (int t = 0; t < 2; t++) {
        v8f z = {};
        z = __builtin_amdgcn_wmma_f32_16x16x32_bf16(false, qa[i][0].v, false,
                                                    kb[t][0].v, (short)0, z,
                                                    false, false);
        z = __builtin_amdgcn_wmma_f32_16x16x32_bf16(false, qa[i][1].v, false,
                                                    kb[t][1].v, (short)0, z,
                                                    false, false);
        s[i][t] = z;
      }

    // ---- ALiBi bias + online softmax (row reductions within 16-lane halves) ----
    #pragma unroll
    for (int i = 0; i < 2; i++)
      #pragma unroll
      for (int r = 0; r < 8; r++) {
        int Mr = r + hiHalf * 8;
        float rowi = (float)(q0 + i * 16 + Mr);
        float c0   = (float)(j0 + lr);
        float s0 = s[i][0][r] + slope * (c0 - rowi);
        float s1 = s[i][1][r] + slope * (c0 + 16.0f - rowi);
        float smax = fmaxf(s0, s1);
        #pragma unroll
        for (int msk = 1; msk < 16; msk <<= 1)
          smax = fmaxf(smax, __shfl_xor(smax, msk, 32));
        float mnew = fmaxf(mrow[i][r], smax);
        float corr = __expf(mrow[i][r] - mnew);
        mrow[i][r] = mnew;
        float p0 = __expf(s0 - mnew);
        float p1 = __expf(s1 - mnew);
        float ps = p0 + p1;
        #pragma unroll
        for (int msk = 1; msk < 16; msk <<= 1)
          ps += __shfl_xor(ps, msk, 32);
        lrow[i][r] = lrow[i][r] * corr + ps;
        #pragma unroll
        for (int c = 0; c < 4; c++) o[i][c][r] *= corr;
        ptile[w][i * 16 + Mr][lr]      = f2bf(p0);
        ptile[w][i * 16 + Mr][lr + 16] = f2bf(p1);
      }
    // per-wave LDS round trip (C-layout -> A-layout); same-wave DS ops are
    // in-order, wait for completion before reading other lanes' data
    asm volatile("s_wait_dscnt 0" ::: "memory");

    BF16Frag pa[2];   // P as A 16x32 over the 32 keys of this block, per row tile
    #pragma unroll
    for (int i = 0; i < 2; i++) {
      const unsigned short* pp = &ptile[w][i * 16 + lr][hiHalf * 16];
      pa[i].h[0] = *(const v8bf*)(pp);
      pa[i].h[1] = *(const v8bf*)(pp + 8);
    }

    // ---- V B-fragments (shared by both row tiles), then O += P.V ----
    BF16Frag vb[4];
    #pragma unroll
    for (int c = 0; c < 4; c++) {
      const unsigned short* vp =
          vbase + (size_t)(c * 16 + lr) * 2048 + j0 + hiHalf * 16;
      vb[c].v = *(const v16bf*)vp;
    }
    #pragma unroll
    for (int i = 0; i < 2; i++)
      #pragma unroll
      for (int c = 0; c < 4; c++)
        o[i][c] = __builtin_amdgcn_wmma_f32_16x16x32_bf16(
            false, pa[i].v, false, vb[c].v, (short)0, o[i][c], false, false);
  }

  // ---- normalize and store fp32 output [B, S, D] ----
  #pragma unroll
  for (int i = 0; i < 2; i++)
    #pragma unroll
    for (int c = 0; c < 4; c++)
      #pragma unroll
      for (int r = 0; r < 8; r++) {
        int Mr = r + hiHalf * 8;
        out[((size_t)(b * 2048 + q0 + i * 16 + Mr)) * 1024 + h * 64 + c * 16 + lr] =
            o[i][c][r] / lrow[i][r];
      }
}

// ---------------------------------------------------------------------------
extern "C" void kernel_launch(void* const* d_in, const int* in_sizes, int n_in,
                              void* d_out, int out_size, void* d_ws, size_t ws_size,
                              hipStream_t stream) {
  const float* x    = (const float*)d_in[0];
  const float* Wkqv = (const float*)d_in[1];
  float* out = (float*)d_out;

  unsigned short* xb = (unsigned short*)d_ws;          // 8192*1024
  unsigned short* wt = xb + (size_t)8192 * 1024;       // 3072*1024
  unsigned short* kq = wt + (size_t)3072 * 1024;       // 8192*2048
  unsigned short* vt = kq + (size_t)8192 * 2048;       // 64*64*2048
  (void)in_sizes; (void)n_in; (void)out_size; (void)ws_size;

  cvt_x_kernel<<<dim3((8192 * 1024) / 256), 256, 0, stream>>>(x, xb, 8192 * 1024);
  cvt_wt_kernel<<<dim3((1024 * 3072) / 256), 256, 0, stream>>>(Wkqv, wt);
  kqv_gemm_kernel<<<dim3(12, 64), 256, 0, stream>>>(xb, wt, kq, vt);   // 128x256 tiles
  alibi_attn_kernel<<<dim3(8, 64), 256, 0, stream>>>(kq, vt, out);     // 256 rows/block
}
